// DCTConv2d_17377437680051
// MI455X (gfx1250) — matrix operand order repaired and checked by reading, not compiled, not run on hardware
//
#include <hip/hip_runtime.h>
#include <hip/hip_bf16.h>
#include <math.h>

typedef _Float16 half_t;
typedef __attribute__((ext_vector_type(16))) _Float16 v16h;
typedef __attribute__((ext_vector_type(4)))  _Float16 v4h;
typedef __attribute__((ext_vector_type(8)))  float    v8f;

// ---------------------------------------------------------------------------
// WMMA fragment helpers (wave32, 16x16x32 f16 -> f32)
// A (16x32): lanes 0-15 hold row M=lane with K in {0..7,16..23};
//            lanes 16-31 hold row M=lane-16 with K in {8..15,24..31}.
// B (32x16): lanes 0-15 hold col N=lane, K in {0..7,16..23}; lanes 16-31 col N=lane-16, K in {8..15,24..31}.
// C/D (16x16 f32): lane<16 -> N=lane, M=r; lane>=16 -> N=lane-16, M=8+r.
// ---------------------------------------------------------------------------

__device__ inline v16h load_a_frag(const half_t* s, int ld, int row0, int k0) {
  int lane = threadIdx.x & 31;
  int m  = lane & 15;
  int ko = (lane >> 4) << 3;                 // 0 or 8
  const half_t* p0 = s + (row0 + m) * ld + k0 + ko;   // K = ko+0..7   (16B contiguous)
  const half_t* p1 = p0 + 16;                          // K = 16+ko+0..7
  v16h a;
#pragma unroll
  for (int h = 0; h < 8; ++h) { a[h] = p0[h]; a[h + 8] = p1[h]; }
  return a;
}

// B fragment from a K-major (transposed) buffer: element B[k][n] stored at s[n*ld + k].
__device__ inline v16h load_b_frag_t(const half_t* s, int ld, int k0, int col0) {
  int lane = threadIdx.x & 31;
  int n  = lane & 15;
  int kb = (lane >> 4) << 3;                 // 0 or 8
  const half_t* p = s + (col0 + n) * ld + k0 + kb;     // K = kb+0..7 and 16+kb+0..7
  v16h b;
#pragma unroll
  for (int h = 0; h < 8; ++h) { b[h] = p[h]; b[h + 8] = p[16 + h]; }
  return b;
}

__device__ inline void store_c_global(float* g, int ld, int row0, int col0, v8f c) {
  int lane = threadIdx.x & 31;
  int n  = lane & 15;
  int mo = (lane >> 4) << 3;
#pragma unroll
  for (int r = 0; r < 8; ++r) g[(size_t)(row0 + mo + r) * ld + col0 + n] = c[r];
}

// Store C transposed ([col][row]) as split hi/lo f16 into LDS.
__device__ inline void store_c_lds_split_t(half_t* sh, half_t* sl, int ld,
                                           int row0, int col0, v8f c) {
  int lane = threadIdx.x & 31;
  int n  = lane & 15;
  int mo = (lane >> 4) << 3;
#pragma unroll
  for (int r = 0; r < 8; ++r) {
    float v = c[r];
    half_t hi = (half_t)v;
    int idx = (col0 + n) * ld + row0 + mo + r;
    sh[idx] = hi;
    sl[idx] = (half_t)(v - (float)hi);
  }
}

// ---------------------------------------------------------------------------
// K0: orthonormal DCT-II table, f16, row-major D[k][m]   (16384 elements)
// ---------------------------------------------------------------------------
__global__ __launch_bounds__(256) void init_dct(half_t* __restrict__ dtab) {
  int idx = blockIdx.x * 256 + threadIdx.x;
  int k = idx >> 7, m = idx & 127;
  float sk  = (k == 0) ? 0.08838834764831845f : 0.125f;  // sqrt(1/128), sqrt(2/128)
  float ang = 3.14159265358979323846f * (2.0f * m + 1.0f) * (float)k / 256.0f;
  dtab[idx] = (half_t)(sk * cosf(ang));
}

// ---------------------------------------------------------------------------
// K1: weights transpose  w[o,i,x,y] -> wt[(x*32+y)*4096 + i*64 + o]  (f16)
// ---------------------------------------------------------------------------
__global__ __launch_bounds__(256) void transpose_w(const float* __restrict__ w,
                                                   half_t* __restrict__ wt) {
  int idx = blockIdx.x * 256 + threadIdx.x;        // 4,194,304 total
  int o = idx & 63;
  int i = (idx >> 6) & 63;
  int y = (idx >> 12) & 31;
  int x = idx >> 17;
  wt[idx] = (half_t)w[(((size_t)o * 64 + i) * 32 + x) * 32 + y];
}

// ---------------------------------------------------------------------------
// K2: fused 2D transform per image:  out = D * in * D^T
// One block (8 waves) per (b,c) image. Split-f16 data, f16 DCT matrix.
// D table is pulled into LDS with CDNA5 async-to-LDS loads (ASYNCcnt),
// overlapped with the f32 -> hi/lo f16 image conversion.
// ---------------------------------------------------------------------------
#define LDP 136
__global__ __launch_bounds__(256) void dct2d_kernel(const float* __restrict__ in,
                                                    float* __restrict__ out,
                                                    const half_t* __restrict__ dtab) {
  extern __shared__ char smem[];
  half_t* sxh = (half_t*)smem;            // [128][LDP] data hi  (x, then T1^T)
  half_t* sxl = sxh + 128 * LDP;          // data lo
  half_t* sD  = sxl + 128 * LDP;          // [128][LDP] D row-major
  const int tid  = threadIdx.x;
  const int wave = tid >> 5;
  const int row0 = wave * 16;
  const size_t base = (size_t)blockIdx.x * (128 * 128);

  // Phase 0a: kick off async LDS fill of the D table (16B chunks, ASYNCcnt).
  {
    unsigned sdbase = (unsigned)(uintptr_t)(void*)sD;
#pragma unroll
    for (int c = 0; c < 8; ++c) {
      int ci = tid + c * 256;                       // 2048 chunks of 16B
      unsigned lds = sdbase + (unsigned)((ci >> 4) * (LDP * 2) + (ci & 15) * 16);
      const void* g = (const void*)(dtab + ci * 8);
      asm volatile("global_load_async_to_lds_b128 %0, %1, off"
                   :: "v"(lds), "v"(g) : "memory");
    }
  }

  // Phase 0b: image -> split f16 LDS (b128 loads, b64 packed stores)
  const float4* in4 = (const float4*)(in + base);
#pragma unroll
  for (int i = 0; i < 16; ++i) {
    int idx = tid + i * 256;                        // 4096 float4
    float4 v = in4[idx];
    int m = idx >> 5;                               // 32 float4 per row
    int n = (idx & 31) * 4;
    half_t h0 = (half_t)v.x, h1 = (half_t)v.y, h2 = (half_t)v.z, h3 = (half_t)v.w;
    *(v4h*)(sxh + m * LDP + n) = (v4h){h0, h1, h2, h3};
    *(v4h*)(sxl + m * LDP + n) =
        (v4h){(half_t)(v.x - (float)h0), (half_t)(v.y - (float)h1),
              (half_t)(v.z - (float)h2), (half_t)(v.w - (float)h3)};
  }
  asm volatile("s_wait_asynccnt 0" ::: "memory");   // D table resident
  __syncthreads();

  // Stage 1: T1[m,l] = sum_n X[m,n] * D[l,n]
  //   A = X (split, rows of this wave's stripe), B[k=n][l] = D[l][n] (K-major in sD rows)
  v8f acc[8] = {};
#pragma unroll
  for (int kk = 0; kk < 128; kk += 32) {
    v16h ah = load_a_frag(sxh, LDP, row0, kk);
    v16h al = load_a_frag(sxl, LDP, row0, kk);
#pragma unroll
    for (int j = 0; j < 8; ++j) {
      v16h b = load_b_frag_t(sD, LDP, kk, j * 16);
      acc[j] = __builtin_amdgcn_wmma_f32_16x16x32_f16(false, ah, false, b,
                                                      (short)0, acc[j], false, false);
      acc[j] = __builtin_amdgcn_wmma_f32_16x16x32_f16(false, al, false, b,
                                                      (short)0, acc[j], false, false);
    }
  }
  __syncthreads();   // all reads of x complete before overwrite
#pragma unroll
  for (int j = 0; j < 8; ++j)   // store T1 transposed+split over the x buffers
    store_c_lds_split_t(sxh, sxl, LDP, row0, j * 16, acc[j]);
  __syncthreads();

  // Stage 2: Z[k,l] = sum_m D[k,m] * T1[m,l]
  //   A = D rows, B[k=m][l] = T1[m,l] stored K-major (split)
  v8f acc2[8] = {};
#pragma unroll
  for (int kk = 0; kk < 128; kk += 32) {
    v16h ad = load_a_frag(sD, LDP, row0, kk);
#pragma unroll
    for (int j = 0; j < 8; ++j) {
      v16h bh = load_b_frag_t(sxh, LDP, kk, j * 16);
      v16h bl = load_b_frag_t(sxl, LDP, kk, j * 16);
      acc2[j] = __builtin_amdgcn_wmma_f32_16x16x32_f16(false, ad, false, bh,
                                                       (short)0, acc2[j], false, false);
      acc2[j] = __builtin_amdgcn_wmma_f32_16x16x32_f16(false, ad, false, bl,
                                                       (short)0, acc2[j], false, false);
    }
  }
#pragma unroll
  for (int j = 0; j < 8; ++j)
    store_c_global(out + base, 128, row0, j * 16, acc2[j]);
}

// ---------------------------------------------------------------------------
// K3: gather corners  z[b,i,row,y] -> zc[(xi*32+y)][b][i]   (LDS-tiled transpose)
// block = (b, xi), xi in [0,64): row = xi<32 ? xi : 96+(xi-32)
// ---------------------------------------------------------------------------
__global__ __launch_bounds__(256) void gather_corners(const float* __restrict__ z,
                                                      float* __restrict__ zc) {
  __shared__ float s[64][33];
  int b   = blockIdx.x >> 6;
  int xi  = blockIdx.x & 63;
  int row = (xi < 32) ? xi : (64 + xi);
  int tid = threadIdx.x;
#pragma unroll
  for (int e = 0; e < 8; ++e) {
    int idx = tid + e * 256;
    int i = idx >> 5, y = idx & 31;
    s[i][y] = z[(((size_t)b * 64 + i) * 128 + row) * 128 + y];
  }
  __syncthreads();
#pragma unroll
  for (int e = 0; e < 8; ++e) {
    int idx = tid + e * 256;
    int y = idx >> 6, i = idx & 63;
    zc[((size_t)(xi * 32 + y)) * 4096 + b * 64 + i] = s[i][y];
  }
}

// ---------------------------------------------------------------------------
// K4: per-position channel mixing, in place:  zc[p][b][o] = sum_i zc[p][b][i]*w[o,i,x,y]
// One block (4 waves) per frequency position p (2048 blocks). 64x64x64 GEMM via WMMA.
// ---------------------------------------------------------------------------
#define LDM 72
__global__ __launch_bounds__(128) void mix_kernel(float* __restrict__ zc,
                                                  const half_t* __restrict__ wt) {
  extern __shared__ char smem[];
  half_t* ah = (half_t*)smem;            // [64][LDM] z hi
  half_t* al = ah + 64 * LDM;            // z lo
  half_t* bw = al + 64 * LDM;            // [o][i] weights, K-major
  int p  = blockIdx.x;                   // 0..2047
  int xi = p >> 5, y = p & 31;
  int x  = (xi < 32) ? xi : (xi - 32);   // both corners share weights
  const half_t* wsrc = wt + ((size_t)(x * 32 + y)) * 4096;   // [i][o]
  float* zp = zc + (size_t)p * 4096;
  int tid = threadIdx.x;

  // z block: b128 loads + packed b64 split stores
  const float4* zp4 = (const float4*)zp;
#pragma unroll
  for (int e = 0; e < 8; ++e) {
    int idx = tid + e * 128;                        // 1024 float4
    float4 v = zp4[idx];
    int r = idx >> 4;                               // 16 float4 per row
    int c = (idx & 15) * 4;
    half_t h0 = (half_t)v.x, h1 = (half_t)v.y, h2 = (half_t)v.z, h3 = (half_t)v.w;
    *(v4h*)(ah + r * LDM + c) = (v4h){h0, h1, h2, h3};
    *(v4h*)(al + r * LDM + c) =
        (v4h){(half_t)(v.x - (float)h0), (half_t)(v.y - (float)h1),
              (half_t)(v.z - (float)h2), (half_t)(v.w - (float)h3)};
  }
  // weights: b32 loads, transpose to [o][i]
  const unsigned* wu = (const unsigned*)wsrc;
#pragma unroll
  for (int e = 0; e < 16; ++e) {
    int idx = tid + e * 128;                        // 2048 uints
    union { unsigned u; half_t h[2]; } cv;
    cv.u = wu[idx];
    int i  = idx >> 5;                              // 32 uints per i-row
    int o0 = (idx & 31) * 2;
    bw[(o0    ) * LDM + i] = cv.h[0];
    bw[(o0 + 1) * LDM + i] = cv.h[1];
  }
  __syncthreads();

  int wave = tid >> 5;
  int row0 = wave * 16;
  v8f acc[4] = {};
#pragma unroll
  for (int kk = 0; kk < 64; kk += 32) {
    v16h fh = load_a_frag(ah, LDM, row0, kk);
    v16h fl = load_a_frag(al, LDM, row0, kk);
#pragma unroll
    for (int j = 0; j < 4; ++j) {
      v16h bf = load_b_frag_t(bw, LDM, kk, j * 16);
      acc[j] = __builtin_amdgcn_wmma_f32_16x16x32_f16(false, fh, false, bf,
                                                      (short)0, acc[j], false, false);
      acc[j] = __builtin_amdgcn_wmma_f32_16x16x32_f16(false, fl, false, bf,
                                                      (short)0, acc[j], false, false);
    }
  }
#pragma unroll
  for (int j = 0; j < 4; ++j)
    store_c_global(zp, 64, row0, j * 16, acc[j]);
}

// ---------------------------------------------------------------------------
// K5: scatter mixed corners back:  zc[(xi*32+y)][b][o] -> z[b,o,row,y]
// ---------------------------------------------------------------------------
__global__ __launch_bounds__(256) void scatter_corners(const float* __restrict__ zc,
                                                       float* __restrict__ z) {
  __shared__ float s[64][33];
  int b   = blockIdx.x >> 6;
  int xi  = blockIdx.x & 63;
  int row = (xi < 32) ? xi : (64 + xi);
  int tid = threadIdx.x;
#pragma unroll
  for (int e = 0; e < 8; ++e) {
    int idx = tid + e * 256;
    int y = idx >> 6, o = idx & 63;
    s[o][y] = zc[((size_t)(xi * 32 + y)) * 4096 + b * 64 + o];
  }
  __syncthreads();
#pragma unroll
  for (int e = 0; e < 8; ++e) {
    int idx = tid + e * 256;
    int o = idx >> 5, y = idx & 31;
    z[(((size_t)b * 64 + o) * 128 + row) * 128 + y] = s[o][y];
  }
}

// ---------------------------------------------------------------------------
extern "C" void kernel_launch(void* const* d_in, const int* in_sizes, int n_in,
                              void* d_out, int out_size, void* d_ws, size_t ws_size,
                              hipStream_t stream) {
  (void)in_sizes; (void)n_in; (void)out_size; (void)ws_size;
  const float* x  = (const float*)d_in[0];   // (64,64,128,128) f32
  const float* w1 = (const float*)d_in[1];   // (64,64,32,32)   f32
  float* out = (float*)d_out;                // (64,64,128,128) f32

  char*   ws   = (char*)d_ws;
  half_t* dtab = (half_t*)ws;                            //  32 KB
  half_t* wt   = (half_t*)(ws + 32768);                  //   8 MB
  float*  zc   = (float*)(ws + 32768 + 8388608);         //  32 MB

  init_dct   <<<64,    256, 0, stream>>>(dtab);
  transpose_w<<<16384, 256, 0, stream>>>(w1, wt);

  const size_t shA = (size_t)3 * 128 * LDP * sizeof(half_t);   // ~102 KB
  dct2d_kernel<<<4096, 256, shA, stream>>>(x, out, dtab);      // forward: z -> d_out

  gather_corners<<<4096, 256, 0, stream>>>(out, zc);

  const size_t shM = (size_t)3 * 64 * LDM * sizeof(half_t);    // ~27 KB
  mix_kernel<<<2048, 128, shM, stream>>>(zc, wt);

  scatter_corners<<<4096, 256, 0, stream>>>(zc, out);

  dct2d_kernel<<<4096, 256, shA, stream>>>(out, out, dtab);    // "inverse" (same op), in place
}
